// InICENODE_78907139162549
// MI455X (gfx1250) — matrix operand order repaired and checked by reading, not compiled, not run on hardware
//
#include <hip/hip_runtime.h>
#include <hip/hip_bf16.h>
#include <math.h>

typedef __attribute__((ext_vector_type(16))) __bf16 v16bf;
typedef __attribute__((ext_vector_type(8)))  float  v8f;

#define NBATCH 1024
#define NSEG   4
#define NTIME  4

#define AS1Q __attribute__((address_space(1)))
#define AS3Q __attribute__((address_space(3)))

// ---------------------------------------------------------------------------
// Address-space helpers: force ds_* for LDS and global_* for device memory.
// Generic->LDS is defined as low-32-bit truncation on AMDGPU (aperture rule:
// LDS address is the low 32 bits); generic->global keeps the 64-bit address.
// ---------------------------------------------------------------------------
static __device__ __forceinline__ unsigned long long p2u(const void* p) {
  return reinterpret_cast<unsigned long long>(p);
}
static __device__ __forceinline__ const AS1Q unsigned int* g_u32(const void* p) {
  return reinterpret_cast<const AS1Q unsigned int*>(p2u(p));
}
static __device__ __forceinline__ const AS3Q unsigned int* l_u32(const void* p) {
  return reinterpret_cast<const AS3Q unsigned int*>((unsigned int)p2u(p));
}
static __device__ __forceinline__ AS3Q float* l_f32(const void* p) {
  return reinterpret_cast<AS3Q float*>((unsigned int)p2u(p));
}
static __device__ __forceinline__ AS3Q __bf16* l_bf16(const void* p) {
  return reinterpret_cast<AS3Q __bf16*>((unsigned int)p2u(p));
}
template <typename T>
static __device__ __forceinline__ T ldg(const T* p) {
  return *reinterpret_cast<const AS1Q T*>(p2u(p));
}
template <typename T>
static __device__ __forceinline__ void stg(T* p, T v) {
  *reinterpret_cast<AS1Q T*>(p2u(p)) = v;
}
static __device__ __forceinline__ float ldl(const float* p) { return *l_f32(p); }
static __device__ __forceinline__ void  stl(float* p, float v) { *l_f32(p) = v; }
static __device__ __forceinline__ void  stlb(__bf16* p, float v) { *l_bf16(p) = (__bf16)v; }

// ---------------------------------------------------------------------------
// Fast transcendentals: gfx1250 TANH/EXP/RCP are TRANS32 VALU ops that
// co-execute with WMMA; avoid branchy libm expansions.
// ---------------------------------------------------------------------------
static __device__ __forceinline__ float tanh_fast(float x) {
#if __has_builtin(__builtin_amdgcn_tanhf)
  return __builtin_amdgcn_tanhf(x);
#else
  float e = __builtin_amdgcn_exp2f(x * 2.8853900818f);  // 2*log2(e)
  return 1.f - 2.f * __builtin_amdgcn_rcpf(e + 1.f);
#endif
}
static __device__ __forceinline__ float sigmoid_fast(float x) {
  return __builtin_amdgcn_rcpf(1.f + __builtin_amdgcn_exp2f(-1.4426950409f * x));
}

// ---------------------------------------------------------------------------
// WMMA fragment helpers (wave32, 16x16x32 bf16, f32 accumulate)
// A 16x32: lane m / m+16 -> row m; vgpr i -> K = (lane/16)*16 + 2i
// B 32x16: lane n / n+16 -> col n; vgpr i -> K = (lane/16)*16 + 2i
// C/D f32: lane -> col (lane&15), vgpr r -> row ((lane>>4)*8 + r)
// ---------------------------------------------------------------------------
static __device__ __forceinline__ v16bf lds_a_frag(const __bf16* base, int ldw, int kt) {
  int lane = threadIdx.x & 31;
  int m  = lane & 15;
  int hi = lane >> 4;
  const AS3Q unsigned int* row = l_u32(base + m * ldw + kt * 32);
  union { v16bf v; unsigned int u[8]; } r;
#pragma unroll
  for (int i = 0; i < 8; i++) {
    int p = (hi << 3) + i;  // 32B contiguous -> ds_load_b128
    r.u[i] = row[p];
  }
  return r.v;
}

static __device__ __forceinline__ v16bf w_frag(const __bf16* w, int kp, int nt, int kt) {
  int lane = threadIdx.x & 31;
  int n  = lane & 15;
  int hi = lane >> 4;
  const AS1Q unsigned int* row = g_u32(w + (size_t)(nt * 16 + n) * kp + kt * 32 + hi * 16);
  union { v16bf v; unsigned int u[8]; } r;
#pragma unroll
  for (int i = 0; i < 8; i++) r.u[i] = row[i];  // 32B contiguous -> 2x global_load_b128
  return r.v;
}

static __device__ __forceinline__ v8f wmma_bf16(v16bf a, v16bf b, v8f c) {
  return __builtin_amdgcn_wmma_f32_16x16x32_bf16(false, a, false, b, (short)0, c, false, false);
}

// act: 0 = linear, 1 = relu, 2 = tanh
static __device__ __forceinline__ void store_tile_bf16(__bf16* out, int ldo, int nt, int N,
                                                       const float* bias, int act, v8f c) {
  int lane = threadIdx.x & 31;
  int n0   = nt * 16 + (lane & 15);
  float bv = (n0 < N) ? ldg(bias + n0) : 0.f;
#pragma unroll
  for (int r = 0; r < 8; r++) {
    int m   = ((lane >> 4) << 3) + r;
    float v = 0.f;
    if (n0 < N) {
      v = c[r] + bv;
      if (act == 1) v = fmaxf(v, 0.f);
      else if (act == 2) v = tanh_fast(v);
    }
    stlb(out + m * ldo + n0, v);  // zero-fills padding columns N..ntiles*16
  }
}

// LDS bf16 -> LDS bf16 GEMM tile: 16 x N += (16 x kp) * (kp x N); 8 waves,
// two independent accumulator chains per wave to cover the WMMA RAW hazard.
static __device__ __forceinline__ void gemm_l2l(const __bf16* in, int ldi, __bf16* out, int ldo,
                                                const __bf16* w, int kp, const float* bias,
                                                int N, int act) {
  int wave   = threadIdx.x >> 5;
  int ntiles = (N + 15) >> 4;
  int ktiles = kp >> 5;
  for (int nt = wave * 2; nt < ntiles; nt += 16) {
    bool two = (nt + 1) < ntiles;
    v8f c0 = {}, c1 = {};
    for (int kt = 0; kt < ktiles; kt++) {
      v16bf a  = lds_a_frag(in, ldi, kt);
      v16bf b0 = w_frag(w, kp, nt, kt);
      c0 = wmma_bf16(a, b0, c0);
      if (two) {
        v16bf b1 = w_frag(w, kp, nt + 1, kt);
        c1 = wmma_bf16(a, b1, c1);
      }
    }
    store_tile_bf16(out, ldo, nt, N, bias, act, c0);
    if (two) store_tile_bf16(out, ldo, nt + 1, N, bias, act, c1);
  }
}

// LDS bf16 -> f32 (LDS and/or global) GEMM tile
static __device__ __forceinline__ void gemm_f32out(const __bf16* in, int ldi, const __bf16* w,
                                                   int kp, const float* bias, int N,
                                                   float* ldsout, int ldl_, float* gout, int grs,
                                                   int act) {
  int wave   = threadIdx.x >> 5;
  int lane   = threadIdx.x & 31;
  int ntiles = (N + 15) >> 4;
  int ktiles = kp >> 5;
  for (int nt = wave; nt < ntiles; nt += 8) {
    v8f c = {};
    for (int kt = 0; kt < ktiles; kt++)
      c = wmma_bf16(lds_a_frag(in, ldi, kt), w_frag(w, kp, nt, kt), c);
    int n0 = nt * 16 + (lane & 15);
    if (n0 < N) {
      float bv = ldg(bias + n0);
#pragma unroll
      for (int r = 0; r < 8; r++) {
        int m   = ((lane >> 4) << 3) + r;
        float v = c[r] + bv;
        if (act == 1) v = fmaxf(v, 0.f);
        if (ldsout) stl(ldsout + m * ldl_ + n0, v);
        if (gout) stg(gout + m * grs + n0, v);
      }
    }
  }
}

// ---------------------------------------------------------------------------
// Main recurrent kernel: one 16-row batch tile per workgroup (batch is the
// only parallel dim; 64 WGs x 8 wave32 each)
// ---------------------------------------------------------------------------
struct MainArgs {
  const float* dxemb;      // [B][32] (30 used)
  const float* int_e;      // [B*S][16] (15 used)
  const float* obs_times;  // [B][S][T]
  const float* obs_vals;   // [B][S][T][64]
  const int*   obs_mask;   // [B][S][T][64]
  float*       out;        // pred_dx [B][256] ++ pred_obs [B][S][T][64]
  const __bf16 *wdyn1, *wdyn2, *wdyn3, *wdyn4;
  const float  *bdyn1, *bdyn2, *bdyn3, *bdyn4;
  const __bf16 *wobs1, *wobs2;
  const float  *bobs1, *bobs2;
  const __bf16 *wdxd1, *wdxd2;
  const float  *bdxd1, *bdxd2;
  const __bf16 *wwi, *wwh;
  const float  *gbi, *gbh;
};

__global__ __launch_bounds__(256) void icenode_main(MainArgs A) {
  __shared__ __align__(16) __bf16 shA[16 * 352];
  __shared__ __align__(16) __bf16 shB[16 * 352];
  __shared__ __align__(16) float shState[16 * 72];
  __shared__ __align__(16) float shKacc[16 * 72];
  __shared__ __align__(16) float shKtmp[16 * 72];
  __shared__ __align__(16) float shCtrl[16 * 16];
  __shared__ __align__(16) float shPred[16 * 64];
  __shared__ __align__(16) float shSum01[16 * 144];
  __shared__ __align__(16) float shInn[16 * 72];
  __shared__ __align__(16) float shHn[16 * 72];
  __shared__ float shH[16];
  __shared__ float shTprev[16];

  const int tid  = threadIdx.x;
  const int row0 = blockIdx.x * 16;

  // one evaluation of the dyn MLP: ktmp = f(state + coeff*h*ktmp, ctrl)
  auto dyn_eval = [&](bool use_k, float coeff) {
    for (int idx = tid; idx < 16 * 96; idx += 256) {
      int m = idx / 96, k = idx - m * 96;
      float v;
      if (k < 70) {
        v = ldl(shState + m * 72 + k);
        if (use_k) v += coeff * ldl(shH + m) * ldl(shKtmp + m * 72 + k);
      } else if (k < 85) {
        v = ldl(shCtrl + m * 16 + (k - 70));
      } else {
        v = 0.f;
      }
      stlb(shA + m * 352 + k, v);
    }
    __syncthreads();
    gemm_l2l(shA, 352, shB, 352, A.wdyn1, 96,  A.bdyn1, 350, 2); __syncthreads();
    gemm_l2l(shB, 352, shA, 352, A.wdyn2, 352, A.bdyn2, 350, 2); __syncthreads();
    gemm_l2l(shA, 352, shB, 352, A.wdyn3, 352, A.bdyn3, 350, 2); __syncthreads();
    gemm_f32out(shB, 352, A.wdyn4, 352, A.bdyn4, 70, shKtmp, 72, nullptr, 0, 0);
    __syncthreads();
  };

  auto rk4_step = [&]() {
#pragma clang loop unroll(disable)
    for (int st = 0; st < 4; st++) {
      float cf = (st == 0) ? 0.f : ((st == 3) ? 1.f : 0.5f);
      float aw = (st == 1 || st == 2) ? 2.f : 1.f;
      dyn_eval(st > 0, cf);
      for (int idx = tid; idx < 16 * 70; idx += 256) {
        int m = idx / 70, j = idx - m * 70;
        float kv = ldl(shKtmp + m * 72 + j);
        float acc = (st == 0) ? kv : ldl(shKacc + m * 72 + j) + aw * kv;
        stl(shKacc + m * 72 + j, acc);
      }
    }
    __syncthreads();
    for (int idx = tid; idx < 16 * 70; idx += 256) {
      int m = idx / 70, j = idx - m * 70;
      stl(shState + m * 72 + j,
          ldl(shState + m * 72 + j) + ldl(shH + m) * (1.f / 6.f) * ldl(shKacc + m * 72 + j));
    }
    __syncthreads();
  };

  // state init: [zeros(40) | dx_emb(30) | pad(2)]
  for (int idx = tid; idx < 16 * 72; idx += 256) {
    int m = idx / 72, j = idx - m * 72;
    float v = 0.f;
    if (j >= 40 && j < 70) v = ldg(A.dxemb + (row0 + m) * 32 + (j - 40));
    stl(shState + idx, v);
  }
  __syncthreads();

#pragma clang loop unroll(disable)
  for (int s = 0; s < NSEG; s++) {
    for (int idx = tid; idx < 16 * 16; idx += 256) {
      int m = idx >> 4, j = idx & 15;
      stl(shCtrl + idx, (j < 15) ? ldg(A.int_e + ((row0 + m) * NSEG + s) * 16 + j) : 0.f);
    }
    if (tid < 16) stl(shTprev + tid, 0.f);
    __syncthreads();

#pragma clang loop unroll(disable)
    for (int t = 0; t <= NTIME; t++) {
      if (tid < 16) {
        float hh;
        if (t < NTIME) {
          float tt = ldg(A.obs_times + ((row0 + tid) * NSEG + s) * NTIME + t);
          hh = (tt - ldl(shTprev + tid)) * (1.f / 24.f);
          stl(shTprev + tid, tt);
        } else {
          hh = (24.f - ldl(shTprev + tid)) * (1.f / 24.f);  // integrate to segment end
        }
        stl(shH + tid, hh);
      }
      __syncthreads();

      rk4_step();
      if (t == NTIME) break;  // uniform: last iteration is the free-run integration only

      // obs decoder: state[15:40] -> 125 relu -> 64
      for (int idx = tid; idx < 16 * 32; idx += 256) {
        int m = idx >> 5, k = idx & 31;
        stlb(shA + m * 352 + k, (k < 25) ? ldl(shState + m * 72 + 15 + k) : 0.f);
      }
      __syncthreads();
      gemm_l2l(shA, 352, shB, 352, A.wobs1, 32, A.bobs1, 125, 1);
      __syncthreads();
      float* gpred = A.out + NBATCH * 256 + (((size_t)row0 * NSEG + s) * NTIME + t) * 64;
      gemm_f32out(shB, 352, A.wobs2, 128, A.bobs2, 64, shPred, 64, gpred, NSEG * NTIME * 64, 0);
      __syncthreads();

      // GRU input: [pred | mask ? val : pred], state as bf16
      for (int idx = tid; idx < 16 * 128; idx += 256) {
        int m = idx >> 7, k = idx & 127;
        float v;
        if (k < 64) {
          v = ldl(shPred + m * 64 + k);
        } else {
          int c  = k - 64;
          int gi = (((row0 + m) * NSEG + s) * NTIME + t) * 64 + c;
          v = ldg(A.obs_mask + gi) ? ldg(A.obs_vals + gi) : ldl(shPred + m * 64 + c);
        }
        stlb(shA + m * 352 + k, v);
      }
      for (int idx = tid; idx < 16 * 96; idx += 256) {
        int m = idx / 96, k = idx - m * 96;
        stlb(shB + m * 352 + k, (k < 70) ? ldl(shState + m * 72 + k) : 0.f);
      }
      __syncthreads();

      // GRU gemms: gi = gru_in @ wi.T (K=128), gh = state @ wh.T (K=96 padded)
      {
        int wave = tid >> 5, lane = tid & 31;
        for (int nt = wave; nt < 14; nt += 8) {
          v8f ci = {}, ch = {};
          for (int kt = 0; kt < 4; kt++)
            ci = wmma_bf16(lds_a_frag(shA, 352, kt), w_frag(A.wwi, 128, nt, kt), ci);
          for (int kt = 0; kt < 3; kt++)
            ch = wmma_bf16(lds_a_frag(shB, 352, kt), w_frag(A.wwh, 96, nt, kt), ch);
          int n0 = nt * 16 + (lane & 15);
          if (n0 < 210) {
            float bvi = ldg(A.gbi + n0), bvh = ldg(A.gbh + n0);
#pragma unroll
            for (int r = 0; r < 8; r++) {
              int m    = ((lane >> 4) << 3) + r;
              float vi = ci[r] + bvi;
              float vh = ch[r] + bvh;
              if (n0 < 140) stl(shSum01 + m * 144 + n0, vi + vh);
              else { stl(shInn + m * 72 + n0 - 140, vi); stl(shHn + m * 72 + n0 - 140, vh); }
            }
          }
        }
      }
      __syncthreads();
      for (int idx = tid; idx < 16 * 70; idx += 256) {
        int m = idx / 70, j = idx - m * 70;
        float r  = sigmoid_fast(ldl(shSum01 + m * 144 + j));
        float z  = sigmoid_fast(ldl(shSum01 + m * 144 + 70 + j));
        float n  = tanh_fast(ldl(shInn + m * 72 + j) + r * ldl(shHn + m * 72 + j));
        float h0 = ldl(shState + m * 72 + j);
        stl(shState + m * 72 + j, (1.f - z) * n + z * h0);
      }
      __syncthreads();
    }
  }

  // dx decoder: state[40:70] -> 150 relu -> 256 -> d_out
  for (int idx = tid; idx < 16 * 32; idx += 256) {
    int m = idx >> 5, k = idx & 31;
    stlb(shA + m * 352 + k, (k < 30) ? ldl(shState + m * 72 + 40 + k) : 0.f);
  }
  __syncthreads();
  gemm_l2l(shA, 352, shB, 352, A.wdxd1, 32, A.bdxd1, 150, 1);
  __syncthreads();
  gemm_f32out(shB, 352, A.wdxd2, 160, A.bdxd2, 256, nullptr, 0, A.out + (size_t)row0 * 256, 256, 0);
}

// ---------------------------------------------------------------------------
// Embedding-phase kernels
// ---------------------------------------------------------------------------
__global__ __launch_bounds__(256) void embed_gemm(const float* Ain, int lda, const __bf16* W,
                                                  int kp, const float* bias, float* C, int ldc,
                                                  int coff, int M, int K, int N, int act) {
  __shared__ __align__(16) __bf16 sA[16 * 256];
  int row0 = blockIdx.x * 16;
  int tid  = threadIdx.x;
  for (int idx = tid; idx < 16 * kp; idx += 256) {
    int m = idx / kp, k = idx - m * kp;
    float v = (k < K) ? ldg(Ain + (size_t)(row0 + m) * lda + k) : 0.f;
    stlb(sA + m * kp + k, v);
  }
  __syncthreads();
  int wave = tid >> 5, lane = tid & 31;
  int ntiles = (N + 15) >> 4, ktiles = kp >> 5;
  for (int nt = wave; nt < ntiles; nt += 8) {
    v8f c = {};
    for (int kt = 0; kt < ktiles; kt++)
      c = wmma_bf16(lds_a_frag(sA, kp, kt), w_frag(W, kp, nt, kt), c);
    int n0 = nt * 16 + (lane & 15);
    if (n0 < N) {
      float bv = ldg(bias + n0);
#pragma unroll
      for (int r = 0; r < 8; r++) {
        int m   = ((lane >> 4) << 3) + r;
        float v = c[r] + bv;
        if (act) v = fmaxf(v, 0.f);
        stg(C + (size_t)(row0 + m) * ldc + coff + n0, v);
      }
    }
  }
}

// f32 (dout x din) -> bf16 zero-padded (np x kp)
__global__ void convert_w(const float* w, __bf16* o, int dout, int din, int np, int kp) {
  int idx = blockIdx.x * blockDim.x + threadIdx.x;
  int tot = np * kp;
  if (idx >= tot) return;
  int n = idx / kp, k = idx - n * kp;
  float v = (n < dout && k < din) ? ldg(w + (size_t)n * din + k) : 0.f;
  stg(o + idx, (__bf16)v);
}

// broadcast demo_emb (B x 5, ld 16) into concat cols 20..24 (ld 32) across S
__global__ void bc_demo(const float* dem, float* cc) {
  int idx = blockIdx.x * blockDim.x + threadIdx.x;
  if (idx >= NBATCH * NSEG * 5) return;
  int j = idx % 5, bs = idx / 5, b = bs / NSEG;
  stg(cc + bs * 32 + 20 + j, ldg(dem + b * 16 + j));
}

// ---------------------------------------------------------------------------
// Host: workspace layout, weight conversion, launches
// ---------------------------------------------------------------------------
extern "C" void kernel_launch(void* const* d_in, const int* in_sizes, int n_in,
                              void* d_out, int out_size, void* d_ws, size_t ws_size,
                              hipStream_t stream) {
  (void)in_sizes; (void)n_in; (void)out_size; (void)ws_size;
  auto F = [&](int i) { return (const float*)d_in[i]; };
  unsigned char* ws = (unsigned char*)d_ws;
  size_t off = 0;
  auto alloc = [&](size_t bytes) -> void* {
    off = (off + 255) & ~(size_t)255;
    void* p = ws + off;
    off += bytes;
    return p;
  };

  // weight table: {input idx, dout, din}; params flattened in sorted-dict order
  struct WDesc { int wi, dout, din; };
  const WDesc wd[20] = {
      {7, 25, 10},   {9, 5, 25},      // dem_emb
      {11, 150, 30}, {13, 256, 150},  // dx_dec
      {15, 150, 256},{17, 30, 150},   // dx_emb
      {19, 350, 85}, {21, 350, 350}, {23, 350, 350}, {25, 70, 350},  // dyn
      {27, 50, 128}, {29, 10, 50},    // inp_emb
      {31, 75, 25},  {33, 15, 75},    // int_emb
      {35, 125, 25}, {37, 64, 125},   // obs_dec
      {39, 50, 64},  {41, 10, 50},    // proc_emb
      {45, 210, 70}, {46, 210, 128},  // update.wh, update.wi
  };
  __bf16* wb[20];
  for (int i = 0; i < 20; i++) {
    int np = ((wd[i].dout + 15) / 16) * 16;
    int kp = ((wd[i].din + 31) / 32) * 32;
    wb[i]  = (__bf16*)alloc((size_t)np * kp * sizeof(__bf16));
    int tot = np * kp;
    convert_w<<<(tot + 255) / 256, 256, 0, stream>>>(F(wd[i].wi), wb[i], wd[i].dout, wd[i].din,
                                                     np, kp);
  }

  float* HID  = (float*)alloc((size_t)4096 * 96 * sizeof(float));
  float* DXE  = (float*)alloc((size_t)1024 * 32 * sizeof(float));
  float* DEM  = (float*)alloc((size_t)1024 * 16 * sizeof(float));
  float* CC   = (float*)alloc((size_t)4096 * 32 * sizeof(float));
  float* INTE = (float*)alloc((size_t)4096 * 16 * sizeof(float));

  // dx_emb: 256 -> 150 relu -> 30
  embed_gemm<<<64, 256, 0, stream>>>(F(0), 256, wb[4], 256, F(16), HID, 160, 0, 1024, 256, 150, 1);
  embed_gemm<<<64, 256, 0, stream>>>(HID, 160, wb[5], 160, F(18), DXE, 32, 0, 1024, 150, 30, 0);
  // dem_emb: 10 -> 25 relu -> 5
  embed_gemm<<<64, 256, 0, stream>>>(F(3), 10, wb[0], 32, F(8), HID, 32, 0, 1024, 10, 25, 1);
  embed_gemm<<<64, 256, 0, stream>>>(HID, 32, wb[1], 32, F(10), DEM, 16, 0, 1024, 25, 5, 0);
  // inp_emb: 128 -> 50 relu -> 10 (into concat cols 0..9)
  embed_gemm<<<256, 256, 0, stream>>>(F(1), 128, wb[10], 128, F(28), HID, 64, 0, 4096, 128, 50, 1);
  embed_gemm<<<256, 256, 0, stream>>>(HID, 64, wb[11], 64, F(30), CC, 32, 0, 4096, 50, 10, 0);
  // proc_emb: 64 -> 50 relu -> 10 (into concat cols 10..19)
  embed_gemm<<<256, 256, 0, stream>>>(F(2), 64, wb[16], 64, F(40), HID, 64, 0, 4096, 64, 50, 1);
  embed_gemm<<<256, 256, 0, stream>>>(HID, 64, wb[17], 64, F(42), CC, 32, 10, 4096, 50, 10, 0);
  // demo broadcast into concat cols 20..24
  bc_demo<<<(NBATCH * NSEG * 5 + 255) / 256, 256, 0, stream>>>(DEM, CC);
  // int_emb: 25 -> 75 relu -> 15
  embed_gemm<<<256, 256, 0, stream>>>(CC, 32, wb[12], 32, F(32), HID, 96, 0, 4096, 25, 75, 1);
  embed_gemm<<<256, 256, 0, stream>>>(HID, 96, wb[13], 96, F(34), INTE, 16, 0, 4096, 75, 15, 0);

  MainArgs a;
  a.dxemb = DXE; a.int_e = INTE;
  a.obs_times = F(4); a.obs_vals = F(5); a.obs_mask = (const int*)d_in[6];
  a.out = (float*)d_out;
  a.wdyn1 = wb[6]; a.wdyn2 = wb[7]; a.wdyn3 = wb[8]; a.wdyn4 = wb[9];
  a.bdyn1 = F(20); a.bdyn2 = F(22); a.bdyn3 = F(24); a.bdyn4 = F(26);
  a.wobs1 = wb[14]; a.wobs2 = wb[15]; a.bobs1 = F(36); a.bobs2 = F(38);
  a.wdxd1 = wb[2]; a.wdxd2 = wb[3]; a.bdxd1 = F(12); a.bdxd2 = F(14);
  a.wwi = wb[19]; a.wwh = wb[18]; a.gbi = F(44); a.gbh = F(43);
  icenode_main<<<NBATCH / 16, 256, 0, stream>>>(a);
}
